// MoBABlock_19378892439590
// MI455X (gfx1250) — matrix operand order repaired and checked by reading, not compiled, not run on hardware
//
#include <hip/hip_runtime.h>
#include <math.h>

#define DEVINL __device__ __forceinline__

typedef unsigned short u16;
typedef __attribute__((ext_vector_type(16))) __bf16 v16bf;
typedef __attribute__((ext_vector_type(8)))  float  v8f;
typedef __attribute__((ext_vector_type(4)))  unsigned uv4;
typedef __attribute__((ext_vector_type(8)))  int      iv8;
typedef __attribute__((ext_vector_type(4)))  int      iv4;

static constexpr int Bb  = 2;
static constexpr int Ss  = 4096;
static constexpr int Dd  = 1024;
static constexpr int Hh  = 8;
static constexpr int DH  = 128;
static constexpr int BLK = 512;
static constexpr int NBLK = 8;
static constexpr int BS  = Bb * Ss;           // 8192 rows
static constexpr float SCALE = 0.08838834764831845f; // 1/sqrt(128)
static constexpr float NEGV  = -1e9f;

// ---------------- bf16 helpers ----------------
DEVINL u16 f2bf(float f) {
  unsigned u = __float_as_uint(f);
  return (u16)((u + 0x7FFFu + ((u >> 16) & 1u)) >> 16);
}
DEVINL float bf2f(u16 h) { return __uint_as_float(((unsigned)h) << 16); }

union FragBF { unsigned u[8]; v16bf v; };

DEVINL v8f wmma_bf16(const FragBF& a, const FragBF& b, v8f c) {
  return __builtin_amdgcn_wmma_f32_16x16x32_bf16(false, a.v, false, b.v,
                                                 (short)0, c, false, false);
}
// A-fragment (16x32 bf16): lane row = lane%16; K index for packed pair i:
DEVINL int a_kidx(int i, int half) { return ((i >> 2) << 4) + half * 8 + ((i & 3) << 1); }
// B-fragment (32x16 bf16): lane col = lane%16; K index for packed pair i:
DEVINL int b_kidx(int i, int half) { return ((i >> 2) << 4) + half * 8 + ((i & 3) << 1); }

// ---------------- Tensor Data Mover: 2D u16 tile -> LDS with row padding ----
// tile_d0 elements contiguous (dim0), tile_d1 rows, row stride stride_d0 (elems).
// pad_interval_code: insert (pad_amount_code+1) DWORDs after every 2^(c+1) DWORDs.
DEVINL void tdm_load_tile_u16(unsigned lds_off, const void* gptr,
                              unsigned tile_d0, unsigned tile_d1,
                              unsigned stride_d0,
                              unsigned pad_interval_code, unsigned pad_amount_code) {
  unsigned long long ga = (unsigned long long)(size_t)gptr;
  uv4 g0;
  g0[0] = 1u;                                   // count=1, user descriptor
  g0[1] = lds_off;                              // LDS byte address
  g0[2] = (unsigned)(ga & 0xFFFFFFFFu);         // global addr [31:0]
  g0[3] = (unsigned)((ga >> 32) & 0x01FFFFFFu) | (2u << 30);  // addr[56:32] | type=2
  iv8 g1;
  g1[0] = (int)((1u << 16)                      // data_size = 1 -> 2 bytes
              | (1u << 20)                      // pad_enable
              | (pad_interval_code << 22)
              | (pad_amount_code << 25));
  g1[1] = (int)((tile_d0 & 0xFFFFu) << 16);     // tensor_dim0[15:0] (== tile_d0)
  g1[2] = (int)((tile_d0 >> 16) | ((tile_d1 & 0xFFFFu) << 16)); // dim0 hi | tensor_dim1 lo
  g1[3] = (int)((tile_d1 >> 16) | ((tile_d0 & 0xFFFFu) << 16)); // dim1 hi | tile_dim0
  g1[4] = (int)(tile_d1 & 0xFFFFu);             // tile_dim1 (tile_dim2 = 0)
  g1[5] = (int)stride_d0;                       // tensor_dim0_stride[31:0]
  g1[6] = 0;
  g1[7] = 0;
  iv4 z4; z4[0] = 0; z4[1] = 0; z4[2] = 0; z4[3] = 0;
  iv8 z8;
#pragma unroll
  for (int j = 0; j < 8; ++j) z8[j] = 0;
  __builtin_amdgcn_tensor_load_to_lds(g0, g1, z4, z4, z8, 0);
}
DEVINL unsigned lds_addr_of(const void* p) { return (unsigned)(size_t)p; }

// ---------------- weight conversion ----------------
__global__ void cvt_f32_bf16(const float* __restrict__ src, u16* __restrict__ dst, int n) {
  int i = blockIdx.x * blockDim.x + threadIdx.x;
  if (i < n) dst[i] = f2bf(src[i]);
}

// ---------------- layernorm (row of 1024) -> bf16 ----------------
__global__ __launch_bounds__(256) void ln_kernel(const float* __restrict__ x,
                                                 const float* __restrict__ w,
                                                 const float* __restrict__ b,
                                                 u16* __restrict__ y) {
  __shared__ float red[16];
  __shared__ float mu_s, ri_s;
  int row = blockIdx.x, tid = threadIdx.x;
  int wave = tid >> 5, lane = tid & 31;
  const float* xr = x + (size_t)row * Dd;
  float vals[4], s = 0.f, ss = 0.f;
#pragma unroll
  for (int j = 0; j < 4; ++j) {
    float v = xr[tid + j * 256];
    vals[j] = v; s += v; ss += v * v;
  }
#pragma unroll
  for (int off = 16; off >= 1; off >>= 1) {
    s  += __shfl_xor(s, off, 32);
    ss += __shfl_xor(ss, off, 32);
  }
  if (lane == 0) { red[wave * 2] = s; red[wave * 2 + 1] = ss; }
  __syncthreads();
  if (tid == 0) {
    float S = 0.f, SS = 0.f;
    for (int wv = 0; wv < 8; ++wv) { S += red[wv * 2]; SS += red[wv * 2 + 1]; }
    float mu = S * (1.f / Dd);
    float var = SS * (1.f / Dd) - mu * mu;
    mu_s = mu; ri_s = rsqrtf(var + 1e-5f);
  }
  __syncthreads();
  float mu = mu_s, ri = ri_s;
#pragma unroll
  for (int j = 0; j < 4; ++j) {
    int col = tid + j * 256;
    y[(size_t)row * Dd + col] = f2bf((vals[j] - mu) * ri * w[col] + b[col]);
  }
}

// ---------------- generic WMMA GEMM: C = A[M,K](bf16) * W[K,N](bf16) + bias ----------------
// A tile staged by the Tensor Data Mover (TDM) with hardware LDS row padding.
// EPI: 0 = f32 out; 2 = exact GELU -> bf16 out; 3 = +residual(f32) -> f32 out
template <int EPI>
__global__ __launch_bounds__(256) void gemm_bf16(const u16* __restrict__ A,
                                                 const u16* __restrict__ W,
                                                 const float* __restrict__ bias,
                                                 const float* __restrict__ res,
                                                 float* __restrict__ outf,
                                                 u16* __restrict__ outb,
                                                 int M, int N, int K) {
  // As: 128 rows x 32 u16, TDM pads 1 DWORD per 16 DWORDs -> row stride 34 u16
  __shared__ __align__(16) u16 As[128 * 34];
  __shared__ __align__(16) u16 Bs[128 * 36];
  int tid = threadIdx.x;
  int wave = tid >> 5, lane = tid & 31;
  int half = lane >> 4, lr = lane & 15;
  int m0 = blockIdx.x * 128, n0 = blockIdx.y * 128;
  int wm = (wave >> 2) * 64;  // 2 wave-rows
  int wn = (wave & 3) * 32;   // 4 wave-cols
  unsigned as_lds = lds_addr_of(As);
  v8f acc[4][2];
#pragma unroll
  for (int mt = 0; mt < 4; ++mt)
#pragma unroll
    for (int nt = 0; nt < 2; ++nt)
#pragma unroll
      for (int j = 0; j < 8; ++j) acc[mt][nt][j] = 0.f;

  for (int k0 = 0; k0 < K; k0 += 32) {
    __syncthreads();
    // stage A tile (128x32) via TDM: tile_d0=32 contiguous, 128 rows, stride K
    if (wave == 0) {
      tdm_load_tile_u16(as_lds, A + (size_t)m0 * K + k0,
                        /*tile_d0=*/32, /*tile_d1=*/128, /*stride=*/(unsigned)K,
                        /*pad_interval(16 dw)=*/3, /*pad_amount(1 dw)=*/0);
    }
    // stage W tile (32x128) transposed -> Bs[n][k]
#pragma unroll
    for (int j = 0; j < 16; ++j) {
      int lin = tid * 16 + j;         // 4096 ushorts
      int kk = lin >> 7, n = lin & 127;
      Bs[n * 36 + kk] = W[(size_t)(k0 + kk) * N + n0 + n];
    }
    if (wave == 0) __builtin_amdgcn_s_wait_tensorcnt(0);
    __syncthreads();
    FragBF bfr[2];
#pragma unroll
    for (int nt = 0; nt < 2; ++nt)
#pragma unroll
      for (int i = 0; i < 8; ++i)
        bfr[nt].u[i] = *(const unsigned*)(Bs + (wn + nt * 16 + lr) * 36 + b_kidx(i, half));
#pragma unroll
    for (int mt = 0; mt < 4; ++mt) {
      FragBF afr;
#pragma unroll
      for (int i = 0; i < 8; ++i)
        afr.u[i] = *(const unsigned*)(As + (wm + mt * 16 + lr) * 34 + a_kidx(i, half));
#pragma unroll
      for (int nt = 0; nt < 2; ++nt) acc[mt][nt] = wmma_bf16(afr, bfr[nt], acc[mt][nt]);
    }
  }
  // epilogue (C layout: row = r + 8*half, col = lr)
#pragma unroll
  for (int mt = 0; mt < 4; ++mt)
#pragma unroll
    for (int nt = 0; nt < 2; ++nt)
#pragma unroll
      for (int r = 0; r < 8; ++r) {
        int row = m0 + wm + mt * 16 + r + 8 * half;
        int col = n0 + wn + nt * 16 + lr;
        float v = acc[mt][nt][r] + bias[col];
        size_t o = (size_t)row * N + col;
        if (EPI == 0) {
          outf[o] = v;
        } else if (EPI == 2) {
          float g = 0.5f * v * (1.f + erff(v * 0.70710678118654752f));
          outb[o] = f2bf(g);
        } else {
          outf[o] = v + res[o];
        }
      }
}

// ---------------- RoPE + head-major relayout -> bf16 ----------------
__global__ __launch_bounds__(256) void rope_kernel(const float* __restrict__ qf,
                                                   const float* __restrict__ kf,
                                                   const float* __restrict__ vf,
                                                   const float* __restrict__ cosT,
                                                   const float* __restrict__ sinT,
                                                   u16* __restrict__ qo,
                                                   u16* __restrict__ ko,
                                                   u16* __restrict__ vo) {
  int bs = blockIdx.x;                  // [0, B*S)
  int b = bs / Ss, s = bs % Ss;
  for (int e = threadIdx.x; e < Dd; e += 256) {
    int h = e >> 7, d = e & 127;
    float c = cosT[s * DH + d], sn = sinT[s * DH + d];
    size_t gin = (size_t)bs * Dd + e;
    float qv = qf[gin], kv = kf[gin], vv = vf[gin];
    float qr, kr;
    if (d < 64) { qr = -qf[gin + 64]; kr = -kf[gin + 64]; }
    else        { qr =  qf[gin - 64]; kr =  kf[gin - 64]; }
    size_t gout = ((size_t)(b * Hh + h) * Ss + s) * DH + d;
    qo[gout] = f2bf(qv * c + qr * sn);
    ko[gout] = f2bf(kv * c + kr * sn);
    vo[gout] = f2bf(vv);
  }
}

// ---------------- block-mean keys ----------------
__global__ __launch_bounds__(128) void kmean_kernel(const u16* __restrict__ ko,
                                                    float* __restrict__ km) {
  int idx = blockIdx.x;                 // B*H*NBLK
  int d = threadIdx.x;                  // 128
  int n = idx & 7, bh = idx >> 3;
  const u16* base = ko + ((size_t)bh * Ss + n * BLK) * DH + d;
  float acc = 0.f;
  for (int j = 0; j < BLK; ++j) acc += bf2f(base[(size_t)j * DH]);
  km[(size_t)idx * DH + d] = acc * (1.f / BLK);
}

// ---------------- gate + top-k selection mask ----------------
__global__ __launch_bounds__(256) void gate_kernel(const u16* __restrict__ qo,
                                                   const float* __restrict__ km,
                                                   unsigned* __restrict__ sel) {
  int wave = threadIdx.x >> 5, lane = threadIdx.x & 31;
  int ridx = blockIdx.x * 8 + wave;     // over B*H*S
  int s = ridx & (Ss - 1), bh = ridx >> 12;
  const u16* qrow = qo + (size_t)ridx * DH + lane * 4;
  float qv[4];
#pragma unroll
  for (int j = 0; j < 4; ++j) qv[j] = bf2f(qrow[j]);
  float dots[NBLK];
#pragma unroll
  for (int n = 0; n < NBLK; ++n) {
    const float* kr = km + ((size_t)bh * NBLK + n) * DH + lane * 4;
    float p = qv[0] * kr[0] + qv[1] * kr[1] + qv[2] * kr[2] + qv[3] * kr[3];
#pragma unroll
    for (int off = 16; off >= 1; off >>= 1) p += __shfl_xor(p, off, 32);
    dots[n] = p;
  }
  if (lane == 0) {
    int vis = s >> 9;
    float sc[NBLK];
#pragma unroll
    for (int n = 0; n < NBLK; ++n) sc[n] = (n <= vis) ? dots[n] * SCALE : NEGV;
    unsigned mask = 1u << (s >> 9);     // own block
#pragma unroll
    for (int t = 0; t < 4; ++t) {
      int bi = 0; float bv = sc[0];
#pragma unroll
      for (int n = 1; n < NBLK; ++n) if (sc[n] > bv) { bv = sc[n]; bi = n; }
      mask |= 1u << bi; sc[bi] = -3e9f;
    }
    sel[ridx] = mask;
  }
}

// ---------------- WMMA flash attention with MoBA block masking ----------------
// K slab staged by TDM (hardware row padding to stride 132), V transposed manually.
__global__ __launch_bounds__(256) void attn_kernel(const u16* __restrict__ qb,
                                                   const u16* __restrict__ kb,
                                                   const u16* __restrict__ vb,
                                                   const unsigned* __restrict__ selmask,
                                                   u16* __restrict__ outb) {
  __shared__ __align__(16) u16 Ks[32 * 132];   // [key][dim], TDM-padded rows
  __shared__ __align__(16) u16 Vs[128 * 34];   // [dim][key] transposed, padded
  __shared__ __align__(16) u16 Ps[8][16 * 36]; // wave-private P tiles
  int bh = blockIdx.x;                  // b*H + h
  int b = bh >> 3, h = bh & 7;
  int q0 = blockIdx.y * 128;
  int tid = threadIdx.x, wave = tid >> 5, lane = tid & 31;
  int half = lane >> 4, lr = lane & 15;
  int qrow0 = q0 + wave * 16;
  const size_t headoff = (size_t)bh * Ss * DH;
  unsigned ks_lds = lds_addr_of(Ks);

  // preload Q A-fragments (4 K-chunks of 32 over DH=128)
  unsigned qa[4][8];
#pragma unroll
  for (int c = 0; c < 4; ++c)
#pragma unroll
    for (int i = 0; i < 8; ++i)
      qa[c][i] = *(const unsigned*)(qb + headoff + (size_t)(qrow0 + lr) * DH +
                                    c * 32 + a_kidx(i, half));
  // per-row selection masks
  unsigned msel = selmask[(size_t)bh * Ss + qrow0 + lr];
  unsigned rm[8], orall = 0;
#pragma unroll
  for (int r = 0; r < 8; ++r) { rm[r] = __shfl(msel, r + 8 * half, 32); orall |= rm[r]; }
  unsigned wsel = orall | __shfl_xor(orall, 16, 32);

  float mrow[8], lrow[8];
  v8f o[8];
#pragma unroll
  for (int r = 0; r < 8; ++r) { mrow[r] = -1e30f; lrow[r] = 0.f; }
#pragma unroll
  for (int vt = 0; vt < 8; ++vt)
#pragma unroll
    for (int j = 0; j < 8; ++j) o[vt][j] = 0.f;

  int ksteps = (q0 + 128) >> 5;
  for (int ks = 0; ks < ksteps; ++ks) {
    int kb32 = ks * 32;
    __syncthreads();
    // stage K slab (32 keys x 128 dims) via TDM with row pad 2 DWORDs / 64 DWORDs
    if (wave == 0) {
      tdm_load_tile_u16(ks_lds, kb + headoff + (size_t)kb32 * DH,
                        /*tile_d0=*/128, /*tile_d1=*/32, /*stride=*/128,
                        /*pad_interval(64 dw)=*/5, /*pad_amount(2 dw)=*/1);
    }
    {   // stage V transposed: Vs[dim][key]
      int key = tid >> 3, cu = tid & 7;
      const u16* srcV = vb + headoff + (size_t)(kb32 + key) * DH + cu * 16;
#pragma unroll
      for (int d = 0; d < 16; ++d) Vs[(cu * 16 + d) * 34 + key] = srcV[d];
    }
    if (wave == 0) __builtin_amdgcn_s_wait_tensorcnt(0);
    __syncthreads();

    bool active = (kb32 <= qrow0 + 15);
    int jb = kb32 >> 9;
    bool blocksel = (wsel >> jb) & 1u;
    if (!(active && blocksel)) continue;

    // scores: two 16-key tiles
    v8f st[2];
#pragma unroll
    for (int t2 = 0; t2 < 2; ++t2) {
#pragma unroll
      for (int j = 0; j < 8; ++j) st[t2][j] = 0.f;
#pragma unroll
      for (int c = 0; c < 4; ++c) {
        FragBF afr, bfr;
#pragma unroll
        for (int i = 0; i < 8; ++i) afr.u[i] = qa[c][i];
#pragma unroll
        for (int i = 0; i < 8; ++i)
          bfr.u[i] = *(const unsigned*)(Ks + (t2 * 16 + lr) * 132 + c * 32 + b_kidx(i, half));
        st[t2] = wmma_bf16(afr, bfr, st[t2]);
      }
    }
    // mask + online softmax + write P to LDS
#pragma unroll
    for (int r = 0; r < 8; ++r) {
      int row = qrow0 + r + 8 * half;
      bool selrow = (rm[r] >> jb) & 1u;
      float s0 = st[0][r] * SCALE, s1 = st[1][r] * SCALE;
      int c0 = kb32 + lr, c1 = kb32 + 16 + lr;
      if (!selrow || c0 > row) s0 = NEGV;
      if (!selrow || c1 > row) s1 = NEGV;
      float mx = fmaxf(s0, s1);
#pragma unroll
      for (int off = 8; off >= 1; off >>= 1) mx = fmaxf(mx, __shfl_xor(mx, off, 32));
      float mnew = fmaxf(mrow[r], mx);
      float p0 = __expf(s0 - mnew), p1 = __expf(s1 - mnew);
      float rs = p0 + p1;
#pragma unroll
      for (int off = 8; off >= 1; off >>= 1) rs += __shfl_xor(rs, off, 32);
      float corr = __expf(mrow[r] - mnew);
      lrow[r] = lrow[r] * corr + rs;
      mrow[r] = mnew;
#pragma unroll
      for (int vt = 0; vt < 8; ++vt) o[vt][r] *= corr;
      Ps[wave][(r + 8 * half) * 36 + lr]      = f2bf(p0);
      Ps[wave][(r + 8 * half) * 36 + 16 + lr] = f2bf(p1);
    }
    asm volatile("s_wait_dscnt 0x0" ::: "memory");  // wave-internal LDS visibility
    // P (16x32) @ V (32x128): 8 output column tiles
    FragBF pa;
#pragma unroll
    for (int i = 0; i < 8; ++i)
      pa.u[i] = *(const unsigned*)(Ps[wave] + lr * 36 + a_kidx(i, half));
#pragma unroll
    for (int vt = 0; vt < 8; ++vt) {
      FragBF vfr;
#pragma unroll
      for (int i = 0; i < 8; ++i)
        vfr.u[i] = *(const unsigned*)(Vs + (vt * 16 + lr) * 34 + b_kidx(i, half));
      o[vt] = wmma_bf16(pa, vfr, o[vt]);
    }
  }
  // finalize: out[b, s, h*128 + dh] (bf16 for the O-projection GEMM)
#pragma unroll
  for (int vt = 0; vt < 8; ++vt)
#pragma unroll
    for (int r = 0; r < 8; ++r) {
      int row = qrow0 + r + 8 * half;
      float val = o[vt][r] / lrow[r];
      outb[((size_t)b * Ss + row) * Dd + h * DH + vt * 16 + lr] = f2bf(val);
    }
}

// ---------------- launch ----------------
extern "C" void kernel_launch(void* const* d_in, const int* in_sizes, int n_in,
                              void* d_out, int out_size, void* d_ws, size_t ws_size,
                              hipStream_t stream) {
  const float* hs    = (const float*)d_in[0];
  const float* ln1_w = (const float*)d_in[1];
  const float* ln1_b = (const float*)d_in[2];
  const float* wq    = (const float*)d_in[3];
  const float* bq    = (const float*)d_in[4];
  const float* wk    = (const float*)d_in[5];
  const float* bk    = (const float*)d_in[6];
  const float* wv    = (const float*)d_in[7];
  const float* bv    = (const float*)d_in[8];
  const float* wo    = (const float*)d_in[9];
  const float* bo    = (const float*)d_in[10];
  const float* ln2_w = (const float*)d_in[11];
  const float* ln2_b = (const float*)d_in[12];
  const float* wi    = (const float*)d_in[13];
  const float* bi    = (const float*)d_in[14];
  const float* wo2   = (const float*)d_in[15];
  const float* bo2   = (const float*)d_in[16];
  const float* cosT  = (const float*)d_in[17];
  const float* sinT  = (const float*)d_in[18];
  float* outp = (float*)d_out;

  // workspace carve-up (bytes, 256-aligned); overlap dead buffers
  char* ws = (char*)d_ws;
  size_t off = 0;
  auto alloc = [&](size_t bytes) { size_t r = off; off = (off + bytes + 255) & ~(size_t)255; return r; };
  const size_t DD2  = (size_t)Dd * Dd * 2;
  const size_t D4D2 = (size_t)Dd * 4 * Dd * 2;
  size_t o_wq  = alloc(DD2),  o_wk = alloc(DD2), o_wv = alloc(DD2), o_wo = alloc(DD2);
  size_t o_wi  = alloc(D4D2), o_wo2 = alloc(D4D2);
  size_t o_xln = alloc((size_t)BS * Dd * 2);          // ln1 out; later reused for ln2 out
  size_t o_qf  = alloc((size_t)BS * Dd * 4);          // qkv f32; region later reused for MLP hidden
  size_t o_kf  = alloc((size_t)BS * Dd * 4);
  size_t o_vf  = alloc((size_t)BS * Dd * 4);
  size_t o_qb  = alloc((size_t)BS * Dd * 2);
  size_t o_kb  = alloc((size_t)BS * Dd * 2);
  size_t o_vb  = alloc((size_t)BS * Dd * 2);
  size_t o_km  = alloc((size_t)Bb * Hh * NBLK * DH * 4);
  size_t o_sel = alloc((size_t)Bb * Hh * Ss * 4);
  size_t o_att = alloc((size_t)BS * Dd * 2);
  size_t o_x1  = alloc((size_t)BS * Dd * 4);
  size_t o_h   = o_qf;                                 // reuse qkv-f32 region (96MB >= 64MB)
  size_t o_xl2 = o_xln;                                // reuse ln buffer

  u16* wqb = (u16*)(ws + o_wq); u16* wkb = (u16*)(ws + o_wk);
  u16* wvb = (u16*)(ws + o_wv); u16* wob = (u16*)(ws + o_wo);
  u16* wib = (u16*)(ws + o_wi); u16* wo2b = (u16*)(ws + o_wo2);
  u16* xln = (u16*)(ws + o_xln);
  float* qf = (float*)(ws + o_qf); float* kf = (float*)(ws + o_kf); float* vf = (float*)(ws + o_vf);
  u16* qbb = (u16*)(ws + o_qb); u16* kbb = (u16*)(ws + o_kb); u16* vbb = (u16*)(ws + o_vb);
  float* km = (float*)(ws + o_km);
  unsigned* sel = (unsigned*)(ws + o_sel);
  u16* attb = (u16*)(ws + o_att);
  float* x1 = (float*)(ws + o_x1);
  u16* hbb = (u16*)(ws + o_h);
  u16* xl2 = (u16*)(ws + o_xl2);

  // 1) weights -> bf16
  int nDD = Dd * Dd, nD4 = Dd * 4 * Dd;
  cvt_f32_bf16<<<(nDD + 255) / 256, 256, 0, stream>>>(wq, wqb, nDD);
  cvt_f32_bf16<<<(nDD + 255) / 256, 256, 0, stream>>>(wk, wkb, nDD);
  cvt_f32_bf16<<<(nDD + 255) / 256, 256, 0, stream>>>(wv, wvb, nDD);
  cvt_f32_bf16<<<(nDD + 255) / 256, 256, 0, stream>>>(wo, wob, nDD);
  cvt_f32_bf16<<<(nD4 + 255) / 256, 256, 0, stream>>>(wi, wib, nD4);
  cvt_f32_bf16<<<(nD4 + 255) / 256, 256, 0, stream>>>(wo2, wo2b, nD4);

  // 2) LN1
  ln_kernel<<<BS, 256, 0, stream>>>(hs, ln1_w, ln1_b, xln);

  // 3) QKV projections (f32 + bias)
  dim3 g1(BS / 128, Dd / 128);
  gemm_bf16<0><<<g1, 256, 0, stream>>>(xln, wqb, bq, nullptr, qf, nullptr, BS, Dd, Dd);
  gemm_bf16<0><<<g1, 256, 0, stream>>>(xln, wkb, bk, nullptr, kf, nullptr, BS, Dd, Dd);
  gemm_bf16<0><<<g1, 256, 0, stream>>>(xln, wvb, bv, nullptr, vf, nullptr, BS, Dd, Dd);

  // 4) RoPE + head-major bf16
  rope_kernel<<<BS, 256, 0, stream>>>(qf, kf, vf, cosT, sinT, qbb, kbb, vbb);

  // 5) gate routing
  kmean_kernel<<<Bb * Hh * NBLK, 128, 0, stream>>>(kbb, km);
  gate_kernel<<<(Bb * Hh * Ss) / 8, 256, 0, stream>>>(qbb, km, sel);

  // 6) MoBA flash attention
  dim3 ga(Bb * Hh, Ss / 128);
  attn_kernel<<<ga, 256, 0, stream>>>(qbb, kbb, vbb, sel, attb);

  // 7) O-projection + residual -> x1
  gemm_bf16<3><<<g1, 256, 0, stream>>>(attb, wob, bo, hs, x1, nullptr, BS, Dd, Dd);

  // 8) LN2
  ln_kernel<<<BS, 256, 0, stream>>>(x1, ln2_w, ln2_b, xl2);

  // 9) MLP in (GELU -> bf16)
  dim3 g2(BS / 128, (4 * Dd) / 128);
  gemm_bf16<2><<<g2, 256, 0, stream>>>(xl2, wib, bi, nullptr, nullptr, hbb, BS, 4 * Dd, Dd);

  // 10) MLP out + residual -> d_out
  gemm_bf16<3><<<g1, 256, 0, stream>>>(hbb, wo2b, bo2, x1, outp, nullptr, BS, Dd, 4 * Dd);
}
